// GLASimple_5909874999710
// MI455X (gfx1250) — matrix operand order repaired and checked by reading, not compiled, not run on hardware
//
#include <hip/hip_runtime.h>
#include <math.h>

// ---------------- types ----------------
typedef __attribute__((ext_vector_type(16))) __bf16        v16bf;
typedef __attribute__((ext_vector_type(8)))  float         v8f;
typedef __attribute__((ext_vector_type(4)))  float         v4f;
typedef __attribute__((ext_vector_type(2)))  float         v2f;
typedef __attribute__((ext_vector_type(4)))  unsigned int  v4u;

#if defined(__AMDGCN__) && __has_builtin(__builtin_amdgcn_tensor_load_to_lds) && __has_builtin(__builtin_amdgcn_s_wait_tensorcnt)
#define USE_TDM 1
typedef __attribute__((ext_vector_type(4))) unsigned int u32x4;
typedef __attribute__((ext_vector_type(8))) int          i32x8;
typedef __attribute__((ext_vector_type(4))) int          i32x4;
#else
#define USE_TDM 0
#endif

// ---------------- problem constants ----------------
#define H_   8
#define DK_  128
#define DV_  256
#define D_   2048
#define KD_  1024      // H*DK
#define VD_  2048      // H*DV
#define B_   2
#define T_   4096
#define M_   (B_*T_)   // 8192

__device__ __forceinline__ unsigned short f2bf(float f) {
  unsigned int u = __builtin_bit_cast(unsigned int, f);
  u += 0x7FFFu + ((u >> 16) & 1u);            // round-to-nearest-even
  return (unsigned short)(u >> 16);
}

// ---------------- fp32 -> bf16 convert (4 elems/thread) ----------------
__global__ __launch_bounds__(256) void cvt_f32_bf16(const float* __restrict__ in,
                                                    unsigned short* __restrict__ out,
                                                    int n4) {
  int i = blockIdx.x * 256 + threadIdx.x;
  if (i >= n4) return;
  v4f x = *(const v4f*)(in + (size_t)i * 4);
  unsigned long long p = (unsigned long long)f2bf(x.x)
                       | ((unsigned long long)f2bf(x.y) << 16)
                       | ((unsigned long long)f2bf(x.z) << 32)
                       | ((unsigned long long)f2bf(x.w) << 48);
  *(unsigned long long*)(out + (size_t)i * 4) = p;
}

// ---------------- bf16 GEMM: C[M,N] = alpha * A[M,K] @ B[K,N] ----------------
// 256 threads = 8 waves (wave32), block tile 128x128, K-step 32.
// Wave grid 2(M) x 4(N); each wave: 4 M-subtiles x 2 N-subtiles of 16x16.
// A tile staged to LDS by the Tensor Data Mover (D# descriptor, TENSORcnt);
// B tile staged transposed by vector loads + ds stores (overlapped with TDM).
#define BM 128
#define BN 128
#define BK 32

union FragBF { v16bf v; v4u u[2]; };

__global__ __launch_bounds__(256) void gemm_bf16_f32(
    const unsigned short* __restrict__ A, const unsigned short* __restrict__ Bm,
    float* __restrict__ C, int M, int N, int K, float alpha) {
  __shared__ __align__(16) unsigned short As[BM * BK];   // row-major [m][k]
  __shared__ __align__(16) unsigned short Bt[BN * BK];   // transposed [n][k]
  const int tid  = threadIdx.x;
  const int lane = tid & 31, wave = tid >> 5;
  const int wm = wave >> 2, wn = wave & 3;
  const int ml = lane & 15, lg = lane >> 4;
  const int row0 = blockIdx.y * BM, col0 = blockIdx.x * BN;
  const int brow = tid >> 3, bcol = (tid & 7) << 4;      // B: 16 elems/thread
#if !USE_TDM
  const int arow = tid >> 1, acol = (tid & 1) << 4;      // A: 16 elems/thread
#else
  const unsigned ldsA = (unsigned)(size_t)(void*)&As[0]; // LDS byte address of tile
#endif

  v8f acc[4][2];
  v8f vz = {};
  for (int i = 0; i < 4; ++i) for (int j = 0; j < 2; ++j) acc[i][j] = vz;

  for (int k0 = 0; k0 < K; k0 += BK) {
#if USE_TDM
    if (wave == 0) {
      // ---- Tensor DMA descriptor (D#), 2D tile: BK x BM of 2-byte elements ----
      unsigned long long ga = (unsigned long long)(size_t)A
                            + ((size_t)row0 * (size_t)K + (size_t)k0) * 2ull;
      u32x4 g0 = { 1u,                                   // count=1 (valid user D#)
                   ldsA,                                 // lds_addr (bytes)
                   (unsigned)ga,                         // global_addr[31:0]
                   (unsigned)(ga >> 32) | (2u << 30) };  // addr[56:32] | type=2
      i32x8 g1 = { (int)(1u << 16),                                     // data_size=1 -> 2B
                   (int)((unsigned)K << 16),                            // tensor_dim0 lo
                   (int)(((unsigned)K >> 16) | ((unsigned)M << 16)),    // dim0 hi | dim1 lo
                   (int)(((unsigned)M >> 16) | ((unsigned)BK << 16)),   // dim1 hi | tile_dim0=BK
                   (int)BM,                                             // tile_dim1=BM, tile_dim2=0
                   (int)K,                                              // dim0_stride lo
                   0, 0 };
      i32x4 z4 = {};
#if __clang_major__ >= 23
      i32x8 z8 = {};
      __builtin_amdgcn_tensor_load_to_lds(g0, g1, z4, z4, z8, 0);
#else
      __builtin_amdgcn_tensor_load_to_lds(g0, g1, z4, z4, 0);
#endif
    }
#else
    { // stage A tile (row-major) with vector loads
      const v4u* src = (const v4u*)(A + (size_t)(row0 + arow) * K + k0 + acol);
      *(v4u*)&As[arow * BK + acol]     = src[0];
      *(v4u*)&As[arow * BK + acol + 8] = src[1];
    }
#endif
    { // stage B tile transposed -> [n][k]  (overlaps with TDM transfer)
      const unsigned short* src = Bm + (size_t)(k0 + brow) * N + col0 + bcol;
      if (k0 + BK < K) __builtin_prefetch(src + (size_t)BK * N, 0, 0);
      union { v4u u; unsigned short s[8]; } b0, b1;
      b0.u = *(const v4u*)src;
      b1.u = *(const v4u*)(src + 8);
      #pragma unroll
      for (int i = 0; i < 8; ++i) {
        Bt[(bcol + i)     * BK + brow] = b0.s[i];
        Bt[(bcol + 8 + i) * BK + brow] = b1.s[i];
      }
    }
#if USE_TDM
    if (wave == 0) __builtin_amdgcn_s_wait_tensorcnt(0);  // A tile landed in LDS
#endif
    __syncthreads();

    FragBF a[4], b[2];
    #pragma unroll
    for (int i = 0; i < 4; ++i) {           // A lane layout: m=ml, K halves by lg
      int r = wm * 64 + 16 * i + ml;
      a[i].u[0] = *(const v4u*)&As[r * BK + lg * 8];
      a[i].u[1] = *(const v4u*)&As[r * BK + 16 + lg * 8];
    }
    #pragma unroll
    for (int j = 0; j < 2; ++j) {           // B lane layout: n=ml, K halves by lg
      int c = wn * 32 + 16 * j + ml;
      b[j].u[0] = *(const v4u*)&Bt[c * BK + lg * 8];
      b[j].u[1] = *(const v4u*)&Bt[c * BK + 16 + lg * 8];
    }
    #pragma unroll
    for (int i = 0; i < 4; ++i)
      #pragma unroll
      for (int j = 0; j < 2; ++j)
        acc[i][j] = __builtin_amdgcn_wmma_f32_16x16x32_bf16(
            false, a[i].v, false, b[j].v, (short)0, acc[i][j], false, false);
    __syncthreads();
  }

  #pragma unroll
  for (int i = 0; i < 4; ++i)
    #pragma unroll
    for (int j = 0; j < 2; ++j) {
      int col = col0 + wn * 32 + 16 * j + ml;
      #pragma unroll
      for (int r = 0; r < 8; ++r) {
        int row = row0 + wm * 64 + 16 * i + lg * 8 + r;
        C[(size_t)row * N + col] = alpha * acc[i][j][r];
      }
    }
}

// ---------------- low-rank gate projection ----------------
__global__ __launch_bounds__(256) void gk_lr0_kernel(const float* __restrict__ X,
                                                     const float* __restrict__ W0,
                                                     float* __restrict__ t0) {
  int idx = blockIdx.x * 256 + threadIdx.x;     // M*16 outputs
  int m = idx >> 4, r = idx & 15;
  const float* x = X + (size_t)m * D_;
  float s = 0.f;
  for (int kk = 0; kk < D_; ++kk) s += x[kk] * W0[kk * 16 + r];
  t0[idx] = s;
}

__global__ __launch_bounds__(256) void gk_lr1_kernel(const float* __restrict__ t0,
                                                     const float* __restrict__ W1,
                                                     const float* __restrict__ bias,
                                                     float* __restrict__ gkout) {
  int idx = blockIdx.x * 256 + threadIdx.x;     // M*KD outputs
  int m = idx >> 10, n = idx & 1023;
  float s = bias[n];
  const float* t = t0 + (size_t)m * 16;
  #pragma unroll
  for (int r = 0; r < 16; ++r) s += t[r] * W1[r * KD_ + n];
  float ls = fminf(s, 0.f) - log1pf(__expf(-fabsf(s)));   // stable logsigmoid
  gkout[idx] = ls * (1.0f / 16.0f);                       // / GATE_NORM
}

// ---------------- chunked GLA recurrence (fp32 WMMA 16x16x4) ----------------
// grid: (B*H, DV/32); block: 64 threads (2 waves). Each block owns S[128,32].
#define CH  16
#define DVB 32

__global__ __launch_bounds__(64) void gla_rec_kernel(
    const float* __restrict__ q, const float* __restrict__ k,
    const float* __restrict__ v, const float* __restrict__ gk,
    float* __restrict__ o) {
  __shared__ __align__(16) float S[DK_ * DVB];    // state slice [dk][dv]
  __shared__ __align__(16) float qg[CH * DK_];    // q * exp(A_t)
  __shared__ __align__(16) float kgS[CH * DK_];   // k * exp(-A_t)
  __shared__ __align__(16) float gb[CH * DK_];    // raw gk chunk
  __shared__ __align__(16) float vs[CH * DVB];    // v chunk slice
  __shared__ __align__(16) float Ps[CH * CH];     // masked attention scores
  __shared__ float eT[DK_];                       // exp(A_total)

  const int bh = blockIdx.x;
  const int b = bh >> 3, h = bh & 7;
  const int dv0 = blockIdx.y * DVB;
  const int tid = threadIdx.x;
  const int lane = tid & 31, wave = tid >> 5;
  const int ml = lane & 15, lg = lane >> 4;
  const int nb = wave * 16;                       // each wave owns 16 dv cols
  const v8f vz = {};

  for (int i = tid; i < DK_ * DVB; i += 64) S[i] = 0.f;
  __syncthreads();

  for (int c = 0; c < T_ / CH; ++c) {
    const size_t tbase = (size_t)(b * T_ + c * CH);
    // stage q/k/gk chunk [16,128] and v slice [16,32]
    for (int i = tid; i < CH * DK_ / 4; i += 64) {
      int t = i >> 5; int d4 = (i & 31) << 2;
      size_t ga = ((tbase + t) * H_ + h) * (size_t)DK_ + d4;
      *(v4f*)&qg[t * DK_ + d4]  = *(const v4f*)&q[ga];
      *(v4f*)&kgS[t * DK_ + d4] = *(const v4f*)&k[ga];
      *(v4f*)&gb[t * DK_ + d4]  = *(const v4f*)&gk[ga];
    }
    for (int i = tid; i < CH * DVB / 4; i += 64) {
      int t = i >> 3; int d4 = (i & 7) << 2;
      size_t ga = ((tbase + t) * H_ + h) * (size_t)DV_ + dv0 + d4;
      *(v4f*)&vs[t * DVB + d4] = *(const v4f*)&v[ga];
    }
    __syncthreads();
    // cumulative decay transform along t (per dk column)
    for (int dd = tid; dd < DK_; dd += 64) {
      float run = 0.f;
      for (int t = 0; t < CH; ++t) {
        run += gb[t * DK_ + dd];
        qg[t * DK_ + dd]  *= __expf(run);
        kgS[t * DK_ + dd] *= __expf(-run);
      }
      eT[dd] = __expf(run);
    }
    __syncthreads();
    // P = qg @ kg^T  (wave 0 only; shared via LDS), causal mask
    if (wave == 0) {
      v8f p = vz;
      for (int k0 = 0; k0 < DK_; k0 += 4) {
        v2f a  = *(const v2f*)&qg[ml * DK_ + k0 + lg * 2];
        v2f bb = *(const v2f*)&kgS[ml * DK_ + k0 + lg * 2];
        p = __builtin_amdgcn_wmma_f32_16x16x4_f32(false, a, false, bb, (short)0, p, false, false);
      }
      #pragma unroll
      for (int r = 0; r < 8; ++r) {
        int m = r + lg * 8;
        Ps[m * CH + ml] = (ml <= m) ? p[r] : 0.f;
      }
    }
    __syncthreads();
    // o = mask(P) @ v  +  qg @ S0   (each wave: its 16 dv cols)
    v8f oc = vz;
    for (int k0 = 0; k0 < CH; k0 += 4) {
      v2f a = *(const v2f*)&Ps[ml * CH + k0 + lg * 2];
      v2f bb; bb.x = vs[(k0 + lg * 2) * DVB + nb + ml];
              bb.y = vs[(k0 + lg * 2 + 1) * DVB + nb + ml];
      oc = __builtin_amdgcn_wmma_f32_16x16x4_f32(false, a, false, bb, (short)0, oc, false, false);
    }
    for (int k0 = 0; k0 < DK_; k0 += 4) {
      v2f a = *(const v2f*)&qg[ml * DK_ + k0 + lg * 2];
      v2f bb; bb.x = S[(k0 + lg * 2) * DVB + nb + ml];
              bb.y = S[(k0 + lg * 2 + 1) * DVB + nb + ml];
      oc = __builtin_amdgcn_wmma_f32_16x16x4_f32(false, a, false, bb, (short)0, oc, false, false);
    }
    #pragma unroll
    for (int r = 0; r < 8; ++r) {
      int t = r + lg * 8;
      size_t ga = ((tbase + t) * H_ + h) * (size_t)DV_ + dv0 + nb + ml;
      o[ga] = oc[r];
    }
    // S <- diag(e^{Atot}) S + (kg*e^{Atot})^T @ v
    for (int mi = 0; mi < 8; ++mi) {
      int dkL = mi * 16 + ml;
      float ea = eT[dkL];
      v8f sc = vz;
      for (int k0 = 0; k0 < CH; k0 += 4) {
        v2f a;  a.x = kgS[(k0 + lg * 2) * DK_ + dkL] * ea;
                a.y = kgS[(k0 + lg * 2 + 1) * DK_ + dkL] * ea;
        v2f bb; bb.x = vs[(k0 + lg * 2) * DVB + nb + ml];
                bb.y = vs[(k0 + lg * 2 + 1) * DVB + nb + ml];
        sc = __builtin_amdgcn_wmma_f32_16x16x4_f32(false, a, false, bb, (short)0, sc, false, false);
      }
      #pragma unroll
      for (int r = 0; r < 8; ++r) {
        int dkr = mi * 16 + lg * 8 + r;
        S[dkr * DVB + nb + ml] = S[dkr * DVB + nb + ml] * eT[dkr] + sc[r];
      }
    }
    __syncthreads();
  }
}

// ---------------- fused RMSNorm * w * swish(g), emit bf16 ----------------
__global__ __launch_bounds__(256) void norm_gate_kernel(
    const float* __restrict__ o, const float* __restrict__ g,
    const float* __restrict__ w, unsigned short* __restrict__ outb) {
  int rid = blockIdx.x * 8 + (threadIdx.x >> 5);   // one wave per (b,t,h) row
  int lane = threadIdx.x & 31;
  const float* orow = o + (size_t)rid * DV_;
  const float* grow = g + (size_t)rid * DV_;
  unsigned short* ob = outb + (size_t)rid * DV_;
  float vals[8]; float ss = 0.f;
  #pragma unroll
  for (int i = 0; i < 8; ++i) { float x = orow[lane + 32 * i]; vals[i] = x; ss += x * x; }
  #pragma unroll
  for (int off = 16; off > 0; off >>= 1) ss += __shfl_xor(ss, off, 32);
  float rr = rsqrtf(ss * (1.0f / DV_) + 1e-5f);
  #pragma unroll
  for (int i = 0; i < 8; ++i) {
    int dv = lane + 32 * i;
    float gg = grow[dv];
    float y = vals[i] * rr * w[dv] * (gg / (1.0f + __expf(-gg)));
    ob[dv] = f2bf(y);
  }
}

// ---------------- host launcher ----------------
extern "C" void kernel_launch(void* const* d_in, const int* in_sizes, int n_in,
                              void* d_out, int out_size, void* d_ws, size_t ws_size,
                              hipStream_t stream) {
  (void)in_sizes; (void)n_in; (void)out_size; (void)ws_size;
  const float* X    = (const float*)d_in[0];
  const float* Wq   = (const float*)d_in[1];
  const float* Wk   = (const float*)d_in[2];
  const float* Wv   = (const float*)d_in[3];
  const float* Wg   = (const float*)d_in[4];
  const float* Wgk0 = (const float*)d_in[5];
  const float* Wgk1 = (const float*)d_in[6];
  const float* bgk1 = (const float*)d_in[7];
  const float* gnw  = (const float*)d_in[8];
  const float* Wo   = (const float*)d_in[9];
  float* out = (float*)d_out;

  char* ws = (char*)d_ws;
  size_t off = 0;
  auto alloc = [&](size_t bytes) -> void* {
    void* p = ws + off; off += (bytes + 255) & ~(size_t)255; return p;
  };
  unsigned short* Xb  = (unsigned short*)alloc((size_t)M_ * D_ * 2);
  unsigned short* Wqb = (unsigned short*)alloc((size_t)D_ * KD_ * 2);
  unsigned short* Wkb = (unsigned short*)alloc((size_t)D_ * KD_ * 2);
  unsigned short* Wvb = (unsigned short*)alloc((size_t)D_ * VD_ * 2);
  unsigned short* Wgb = (unsigned short*)alloc((size_t)D_ * VD_ * 2);
  unsigned short* Wob = (unsigned short*)alloc((size_t)VD_ * D_ * 2);
  float* qf  = (float*)alloc((size_t)M_ * KD_ * 4);
  float* kf  = (float*)alloc((size_t)M_ * KD_ * 4);
  float* vf  = (float*)alloc((size_t)M_ * VD_ * 4);
  float* gf  = (float*)alloc((size_t)M_ * VD_ * 4);
  float* t0  = (float*)alloc((size_t)M_ * 16 * 4);
  float* gkf = (float*)alloc((size_t)M_ * KD_ * 4);
  float* of  = (float*)alloc((size_t)M_ * VD_ * 4);
  unsigned short* nb = (unsigned short*)alloc((size_t)M_ * VD_ * 2);

  auto cvt = [&](const float* in, unsigned short* o_, size_t n) {
    int n4 = (int)(n / 4);
    cvt_f32_bf16<<<dim3((n4 + 255) / 256), dim3(256), 0, stream>>>(in, o_, n4);
  };
  cvt(X,  Xb,  (size_t)M_ * D_);
  cvt(Wq, Wqb, (size_t)D_ * KD_);
  cvt(Wk, Wkb, (size_t)D_ * KD_);
  cvt(Wv, Wvb, (size_t)D_ * VD_);
  cvt(Wg, Wgb, (size_t)D_ * VD_);
  cvt(Wo, Wob, (size_t)VD_ * D_);

  const float qscale = 1.0f / sqrtf((float)DK_);
  dim3 blk(256);
  gemm_bf16_f32<<<dim3(KD_ / BN, M_ / BM), blk, 0, stream>>>(Xb, Wqb, qf, M_, KD_, D_, qscale);
  gemm_bf16_f32<<<dim3(KD_ / BN, M_ / BM), blk, 0, stream>>>(Xb, Wkb, kf, M_, KD_, D_, 1.0f);
  gemm_bf16_f32<<<dim3(VD_ / BN, M_ / BM), blk, 0, stream>>>(Xb, Wvb, vf, M_, VD_, D_, 1.0f);
  gemm_bf16_f32<<<dim3(VD_ / BN, M_ / BM), blk, 0, stream>>>(Xb, Wgb, gf, M_, VD_, D_, 1.0f);

  gk_lr0_kernel<<<dim3(M_ * 16 / 256), blk, 0, stream>>>(X, Wgk0, t0);
  gk_lr1_kernel<<<dim3(M_ * KD_ / 256), blk, 0, stream>>>(t0, Wgk1, bgk1, gkf);

  gla_rec_kernel<<<dim3(B_ * H_, DV_ / DVB), dim3(64), 0, stream>>>(qf, kf, vf, gkf, of);

  norm_gate_kernel<<<dim3(M_ * H_ / 8), blk, 0, stream>>>(of, gf, gnw, nb);

  gemm_bf16_f32<<<dim3(D_ / BN, M_ / BM), blk, 0, stream>>>(nb, Wob, out, M_, D_, VD_, 1.0f);
}
